// VQ2_32727650796017
// MI455X (gfx1250) — compile-verified, matrix operand order checked by reading, not run on hardware
//
#include <hip/hip_runtime.h>

#define BT   16384   // 16*1024 rows
#define D    512
#define NE   8192
#define BETA 0.25f

typedef __attribute__((ext_vector_type(8)))  __bf16 v8bf;
typedef __attribute__((ext_vector_type(16))) __bf16 v16bf;
typedef __attribute__((ext_vector_type(8)))  float  v8f;

__device__ __forceinline__ unsigned short f2bf(float f) {
  union { float f; unsigned u; } v; v.f = f;
  unsigned u = v.u;
  return (unsigned short)((u + 0x7FFFu + ((u >> 16) & 1u)) >> 16);
}

// ---- fp32 -> bf16 conversion (stored as u16 bit pattern) ----
__global__ void vq_tobf16(const float* __restrict__ src,
                          unsigned short* __restrict__ dst, int n) {
  int i = blockIdx.x * blockDim.x + threadIdx.x;
  int stride = gridDim.x * blockDim.x;
  for (; i < n; i += stride) dst[i] = f2bf(src[i]);
}

// ---- per-code squared norms: one wave32 per code ----
__global__ void vq_enorm(const float* __restrict__ emb, float* __restrict__ enorm2) {
  int wave = threadIdx.x >> 5;
  int lane = threadIdx.x & 31;
  int c = blockIdx.x * 8 + wave;
  const float* row = emb + (size_t)c * D;
  float s = 0.f;
  for (int j = lane; j < D; j += 32) { float e = row[j]; s += e * e; }
  #pragma unroll
  for (int off = 16; off >= 1; off >>= 1) s += __shfl_down(s, off, 32);
  if (lane == 0) enorm2[c] = s;
}

__global__ void vq_zero3(float* a) { if (threadIdx.x < 3) a[threadIdx.x] = 0.f; }

// ---- fused GEMM + argmin via bf16 WMMA + async LDS staging ----
#define ROWS 128            // z-rows per workgroup (16 per wave, 8 waves)
#define NB   64             // codebook rows per LDS chunk (4 subtiles / wave)
#define A_BYTES (ROWS * D * 2)   // 128 KB
#define B_BYTES (NB * D * 2)     //  64 KB (x2 buffers)

// gfx1250 async memory->LDS copy, GVS form:
//   global_load_async_to_lds_b128 vdst(LDS byte addr), vaddr(32b off), saddr
// Tracked by ASYNCcnt. Dynamic LDS starts at offset 0; HW adds LDS_BASE.
__device__ __forceinline__ void async_ld16(unsigned lds_byte, unsigned voff,
                                           const void* sbase) {
  asm volatile("global_load_async_to_lds_b128 %0, %1, %2"
               :: "v"(lds_byte), "v"(voff), "s"(sbase) : "memory");
}
__device__ __forceinline__ void wait_async0() {
  asm volatile("s_wait_asynccnt 0x0" ::: "memory");
}

__global__ void __launch_bounds__(256) vq_argmin(
    const unsigned short* __restrict__ zb,
    const unsigned short* __restrict__ eb,
    const float* __restrict__ enorm2,
    int* __restrict__ idxOut) {
  extern __shared__ unsigned short smem[];
  unsigned short* sA = smem;               // [0, A_BYTES)
  // B buffers at A_BYTES and A_BYTES + B_BYTES

  const int tid  = threadIdx.x;
  const int wave = tid >> 5;
  const int lane = tid & 31;
  const int g    = lane >> 4;   // half-wave group (selects K sub-blocks)
  const int r    = lane & 15;   // row (A) / column (B) within tile
  const int rowBase = blockIdx.x * ROWS;

  // Async-stage the z tile (contiguous) and the first codebook chunk.
  {
    const void* srcA = (const void*)(zb + (size_t)rowBase * D);
    for (int i = tid; i < A_BYTES / 16; i += 256)
      async_ld16((unsigned)(i * 16), (unsigned)(i * 16), srcA);
    const void* srcB = (const void*)eb;
    for (int i = tid; i < B_BYTES / 16; i += 256)
      async_ld16((unsigned)(A_BYTES + i * 16), (unsigned)(i * 16), srcB);
  }
  wait_async0();
  __syncthreads();

  float bestS[8];
  int   bestI[8];
  #pragma unroll
  for (int v = 0; v < 8; ++v) { bestS[v] = -3.4e38f; bestI[v] = 0; }

  const unsigned short* arow = sA + (size_t)(wave * 16 + r) * D;

  int buf = 0;
  for (int cb = 0; cb < NE; cb += NB) {
    // Prefetch next chunk into the other buffer while we compute (async DMA
    // overlaps the 64 WMMAs below; codebook is L2-resident at 8 MB).
    if (cb + NB < NE) {
      const void* src = (const void*)(eb + (size_t)(cb + NB) * D);
      unsigned base = (unsigned)(A_BYTES + (buf ^ 1) * B_BYTES);
      for (int i = tid; i < B_BYTES / 16; i += 256)
        async_ld16(base + (unsigned)(i * 16), (unsigned)(i * 16), src);
    }

    const unsigned short* sB = smem + (A_BYTES + buf * B_BYTES) / 2;
    v8f acc[4];
    #pragma unroll
    for (int s4 = 0; s4 < 4; ++s4) acc[s4] = (v8f){};

    #pragma unroll
    for (int kc = 0; kc < D / 32; ++kc) {
      // 16-bit A-matrix 16x32 layout: lane group g holds K = g*8..g*8+7
      // (VGPR0-3) and K = 16+g*8..16+g*8+7 (VGPR4-7) of row r.
      const unsigned short* ap = arow + kc * 32 + g * 8;
      v8bf a0 = *(const v8bf*)(ap);
      v8bf a1 = *(const v8bf*)(ap + 16);
      v16bf A = __builtin_shufflevector(a0, a1, 0,1,2,3,4,5,6,7,
                                                8,9,10,11,12,13,14,15);
      #pragma unroll
      for (int s4 = 0; s4 < 4; ++s4) {       // one A-frag feeds 4 WMMAs
        const unsigned short* bp = sB + (size_t)(s4 * 16 + r) * D
                                      + kc * 32 + g * 8;
        v8bf b0 = *(const v8bf*)(bp);
        v8bf b1 = *(const v8bf*)(bp + 16);
        v16bf B = __builtin_shufflevector(b0, b1, 0,1,2,3,4,5,6,7,
                                                  8,9,10,11,12,13,14,15);
        acc[s4] = __builtin_amdgcn_wmma_f32_16x16x32_bf16(
            false, A, false, B, (short)0, acc[s4], false, false);
      }
    }

    // C/D layout: lane holds column n = r, VGPR v holds row v + 8*g.
    #pragma unroll
    for (int s4 = 0; s4 < 4; ++s4) {
      int nc = cb + s4 * 16 + r;
      float en = enorm2[nc];
      #pragma unroll
      for (int v = 0; v < 8; ++v) {
        float score = 2.0f * acc[s4][v] - en;   // argmax == argmin distance
        if (score > bestS[v]) { bestS[v] = score; bestI[v] = nc; }
      }
    }

    wait_async0();       // next chunk's DMA done
    __syncthreads();     // all waves done reading current buffer
    buf ^= 1;
  }

  // Butterfly argmax within each 16-lane half (ties -> smallest index)
  #pragma unroll
  for (int v = 0; v < 8; ++v) {
    float s = bestS[v];
    int bi = bestI[v];
    #pragma unroll
    for (int off = 8; off >= 1; off >>= 1) {
      float so = __shfl_xor(s, off, 32);
      int   io = __shfl_xor(bi, off, 32);
      if (so > s || (so == s && io < bi)) { s = so; bi = io; }
    }
    if (r == 0) idxOut[rowBase + wave * 16 + v + 8 * g] = bi;
  }
}

// ---- gather z_q, write outputs, accumulate loss / idx stats ----
__global__ void __launch_bounds__(256) vq_gather_loss(
    const float* __restrict__ z, const float* __restrict__ emb,
    const int* __restrict__ idx, float* __restrict__ out_zq,
    float* __restrict__ out_idx, float* __restrict__ accum) {
  __shared__ float red[256];
  int row = blockIdx.x;
  int id = idx[row];
  const float* zr = z   + (size_t)row * D;
  const float* er = emb + (size_t)id  * D;
  float* o = out_zq + (size_t)row * D;
  float s = 0.f;
  for (int j = threadIdx.x; j < D; j += 256) {
    float q = er[j];
    float d = q - zr[j];
    s += d * d;
    o[j] = q;                    // straight-through value == z_q
  }
  red[threadIdx.x] = s;
  __syncthreads();
  for (int off = 128; off >= 1; off >>= 1) {
    if (threadIdx.x < off) red[threadIdx.x] += red[threadIdx.x + off];
    __syncthreads();
  }
  if (threadIdx.x == 0) {
    atomicAdd(accum + 0, red[0]);
    float fi = (float)id;
    out_idx[row] = fi;
    atomicAdd(accum + 1, fi);
    atomicAdd(accum + 2, fi * fi);
  }
}

__global__ void vq_finalize(const float* __restrict__ accum,
                            float* loss_out, float* var_out) {
  if (threadIdx.x == 0) {
    float inv = 1.0f / (float)((size_t)BT * D);
    *loss_out = (1.0f + BETA) * accum[0] * inv;   // beta*commit + codebook
    float mean = accum[1] / (float)BT;
    *var_out = accum[2] / (float)BT - mean * mean;
  }
}

extern "C" void kernel_launch(void* const* d_in, const int* in_sizes, int n_in,
                              void* d_out, int out_size, void* d_ws, size_t ws_size,
                              hipStream_t stream) {
  const float* z   = (const float*)d_in[0];
  const float* emb = (const float*)d_in[1];
  float* out = (float*)d_out;

  // workspace layout
  char* ws = (char*)d_ws;
  unsigned short* zb = (unsigned short*)ws;                          // BT*D bf16
  unsigned short* eb = (unsigned short*)(ws + (size_t)BT * D * 2);   // NE*D bf16
  float* enorm2 = (float*)(ws + (size_t)BT * D * 2 + (size_t)NE * D * 2);
  int*   idxbuf = (int*)((char*)enorm2 + (size_t)NE * sizeof(float));
  float* accum  = (float*)((char*)idxbuf + (size_t)BT * sizeof(int));

  // output layout: z_q_st [BT*D] | loss [1] | idx [BT] | var [1]
  float* out_zq   = out;
  float* out_loss = out + (size_t)BT * D;
  float* out_idx  = out + (size_t)BT * D + 1;
  float* out_var  = out + (size_t)BT * D + 1 + BT;

  vq_tobf16<<<4096, 256, 0, stream>>>(z,   zb, BT * D);
  vq_tobf16<<<2048, 256, 0, stream>>>(emb, eb, NE * D);
  vq_enorm <<<NE / 8, 256, 0, stream>>>(emb, enorm2);
  vq_zero3 <<<1, 32, 0, stream>>>(accum);

  size_t smem = (size_t)A_BYTES + 2 * (size_t)B_BYTES;   // 256 KB
  vq_argmin<<<BT / ROWS, 256, smem, stream>>>(zb, eb, enorm2, idxbuf);

  vq_gather_loss<<<BT, 256, 0, stream>>>(z, emb, idxbuf, out_zq, out_idx, accum);
  vq_finalize<<<1, 32, 0, stream>>>(accum, out_loss, out_var);
}
